// KoLeoLoss_46600395161951
// MI455X (gfx1250) — matrix operand order repaired and checked by reading, not compiled, not run on hardware
//
#include <hip/hip_runtime.h>
#include <math.h>

#define A_N 4096
#define C_N 8
#define F_N 64
#define EPSF 1e-6f

typedef __attribute__((ext_vector_type(16))) _Float16 v16h;
typedef __attribute__((ext_vector_type(8)))  float    v8f;

// ------------------------------------------------------------------
// Kernel 1: normalize rows, write f16 normalized data channel-major:
// xn16[c][a][f]. One wave per (a,c) row; F=64 -> 2 elements per lane.
// ------------------------------------------------------------------
__global__ void __launch_bounds__(256) k_normalize(const float* __restrict__ x,
                                                   _Float16* __restrict__ xn16) {
    int wid  = (blockIdx.x * blockDim.x + threadIdx.x) >> 5;
    int lane = threadIdx.x & 31;
    if (wid >= A_N * C_N) return;
    int a = wid / C_N, c = wid % C_N;
    const float* p = x + ((size_t)a * C_N + c) * F_N;
    float v0 = p[lane], v1 = p[lane + 32];
    float ss = v0 * v0 + v1 * v1;
    #pragma unroll
    for (int off = 16; off; off >>= 1) ss += __shfl_xor(ss, off, 32);
    float inv = 1.0f / fmaxf(sqrtf(ss), EPSF);
    _Float16* q = xn16 + ((size_t)c * A_N + a) * F_N;
    q[lane]      = (_Float16)(v0 * inv);
    q[lane + 32] = (_Float16)(v1 * inv);
}

// ------------------------------------------------------------------
// Kernel 2: fused WMMA similarity + per-row argmax.
// Block = 4 waves, all on the SAME channel; each wave owns a 16-row
// A tile. The 4096x64(f16) B panel is streamed through LDS in
// 32-column stages with async global->LDS copies (double-buffered,
// ASYNCcnt + barrier pipelined). Each stage: 4 interleaved
// v_wmma_f32_16x16x32_f16 (two independent accumulator chains).
// ------------------------------------------------------------------
#define STAGE_COLS 32
#define ROW_H      72          // padded LDS row stride in halves (144 B)
#define STAGE_BYTES (STAGE_COLS * ROW_H * 2)

__global__ void __launch_bounds__(128) k_argmax(const _Float16* __restrict__ xn16,
                                                int* __restrict__ nn_idx) {
    __shared__ _Float16 lb[2][STAGE_COLS][ROW_H];

    const int wave = threadIdx.x >> 5;
    const int lane = threadIdx.x & 31;
    const int tile = blockIdx.x * 4 + wave;     // 2048 tiles
    const int c    = tile >> 8;                 // 256 row-tiles / channel
    const int a0   = (tile & 255) * 16;
    const int n    = lane & 15;
    const int half = lane >> 4;

    const _Float16* panel = xn16 + (size_t)c * A_N * F_N;

    // ---- A fragments (16x64), contiguous k = 16*half + j (same order as B) ----
    const _Float16* pa = panel + (size_t)(a0 + n) * F_N;
    v16h af0, af1;
    #pragma unroll
    for (int j = 0; j < 8; ++j) {
        af0[j]     = pa[      16 * half + j];
        af0[8 + j] = pa[8 +  16 * half + j];
        af1[j]     = pa[32 + 16 * half + j];
        af1[8 + j] = pa[40 + 16 * half + j];
    }

    const unsigned lds_base = (unsigned)(uintptr_t)&lb[0][0][0];

    // Async-prefetch one 32-column stage (32 rows x 128 B = 256 chunks of
    // 16 B; 128 threads x 2 chunks). Each wave issues 2 async b128 ops.
    auto prefetch = [&](int buf, int stage) {
        const _Float16* g = panel + (size_t)stage * STAGE_COLS * F_N;
        #pragma unroll
        for (int q = 0; q < 2; ++q) {
            int cid = threadIdx.x + q * 128;
            int row = cid >> 3, k = cid & 7;
            unsigned loff = lds_base + (unsigned)buf * STAGE_BYTES
                          + (unsigned)(row * ROW_H * 2 + k * 16);
            unsigned long long ga =
                (unsigned long long)(uintptr_t)(g + row * F_N + k * 8);
            asm volatile("global_load_async_to_lds_b128 %0, %1, off"
                         :: "v"(loff), "v"(ga) : "memory");
        }
    };

    float best[8];
    int   bidx[8];
    #pragma unroll
    for (int r = 0; r < 8; ++r) { best[r] = -3.0e38f; bidx[r] = 0; }

    prefetch(0, 0);
    asm volatile("s_wait_asynccnt 0" ::: "memory");
    __syncthreads();

    for (int s = 0; s < A_N / STAGE_COLS; ++s) {
        const int cur = s & 1;
        if (s + 1 < A_N / STAGE_COLS) prefetch(cur ^ 1, s + 1);

        // B fragments for the two 16-column tiles of this stage (from LDS).
        v16h b00, b01, b10, b11;
        #pragma unroll
        for (int j = 0; j < 16; ++j) {
            b00[j] = lb[cur][n     ][     16 * half + j];
            b01[j] = lb[cur][n     ][32 + 16 * half + j];
            b10[j] = lb[cur][n + 16][     16 * half + j];
            b11[j] = lb[cur][n + 16][32 + 16 * half + j];
        }
        v8f acc0 = {}, acc1 = {};
        acc0 = __builtin_amdgcn_wmma_f32_16x16x32_f16(false, af0, false, b00,
                                                      (short)0, acc0, false, false);
        acc1 = __builtin_amdgcn_wmma_f32_16x16x32_f16(false, af0, false, b10,
                                                      (short)0, acc1, false, false);
        acc0 = __builtin_amdgcn_wmma_f32_16x16x32_f16(false, af1, false, b01,
                                                      (short)0, acc0, false, false);
        acc1 = __builtin_amdgcn_wmma_f32_16x16x32_f16(false, af1, false, b11,
                                                      (short)0, acc1, false, false);

        const int b0 = s * STAGE_COLS;
        // Diagonal tile: rare (wave-uniform branch, 1 stage out of 128).
        if (b0 == (a0 & ~31)) {
            #pragma unroll
            for (int r = 0; r < 8; ++r) {
                int arow = a0 + r + 8 * half;
                if (b0 + n      == arow) acc0[r] = -1.0f;
                if (b0 + 16 + n == arow) acc1[r] = -1.0f;
            }
        }
        #pragma unroll
        for (int r = 0; r < 8; ++r) {          // strict '>' keeps first max
            float v0 = acc0[r], v1 = acc1[r];
            if (v0 > best[r]) { best[r] = v0; bidx[r] = b0 + n; }
            if (v1 > best[r]) { best[r] = v1; bidx[r] = b0 + 16 + n; }
        }

        asm volatile("s_wait_asynccnt 0" ::: "memory");
        __syncthreads();
    }

    // Cross-lane reduce over the 16 columns in each half; tie-break toward
    // the smaller index to match jnp.argmax first-occurrence semantics.
    #pragma unroll
    for (int r = 0; r < 8; ++r) {
        float bv = best[r]; int bi = bidx[r];
        #pragma unroll
        for (int off = 8; off; off >>= 1) {
            float ov = __shfl_xor(bv, off, 16);
            int   oi = __shfl_xor(bi, off, 16);
            if (ov > bv || (ov == bv && oi < bi)) { bv = ov; bi = oi; }
        }
        if (n == 0) nn_idx[(size_t)c * A_N + a0 + r + 8 * half] = bi;
    }
}

// ------------------------------------------------------------------
// Kernel 3: per-(a,c) distance term in f32 (norms recomputed from x),
// deterministic block tree reduction -> partial sums.
// ------------------------------------------------------------------
__global__ void __launch_bounds__(256) k_loss(const float* __restrict__ x,
                                              const int* __restrict__ nn_idx,
                                              float* __restrict__ partial) {
    int t = blockIdx.x * blockDim.x + threadIdx.x;
    int a = t / C_N, c = t % C_N;
    int idx = nn_idx[(size_t)c * A_N + a];
    const float* xa = x + ((size_t)a   * C_N + c) * F_N;
    const float* xb = x + ((size_t)idx * C_N + c) * F_N;
    float ssa = 0.f, ssb = 0.f;
    #pragma unroll 8
    for (int f = 0; f < F_N; ++f) { ssa += xa[f] * xa[f]; ssb += xb[f] * xb[f]; }
    float ia = 1.0f / fmaxf(sqrtf(ssa), EPSF);
    float ib = 1.0f / fmaxf(sqrtf(ssb), EPSF);
    float d = 0.f;
    #pragma unroll 8
    for (int f = 0; f < F_N; ++f) {
        float u = xa[f] * ia - xb[f] * ib + 1e-6f;   // PDIST_EPS inside norm
        d += u * u;
    }
    float term = logf(sqrtf(d) + EPSF);

    __shared__ float sm[256];
    sm[threadIdx.x] = term;
    __syncthreads();
    #pragma unroll
    for (int s = 128; s; s >>= 1) {
        if (threadIdx.x < s) sm[threadIdx.x] += sm[threadIdx.x + s];
        __syncthreads();
    }
    if (threadIdx.x == 0) partial[blockIdx.x] = sm[0];
}

// ------------------------------------------------------------------
// Kernel 4: final deterministic reduction of 128 partials.
// ------------------------------------------------------------------
__global__ void __launch_bounds__(128) k_final(const float* __restrict__ partial,
                                               float* __restrict__ out) {
    __shared__ float sm[128];
    sm[threadIdx.x] = partial[threadIdx.x];
    __syncthreads();
    #pragma unroll
    for (int s = 64; s; s >>= 1) {
        if (threadIdx.x < s) sm[threadIdx.x] += sm[threadIdx.x + s];
        __syncthreads();
    }
    if (threadIdx.x == 0) out[0] = -sm[0] / (float)(A_N * C_N);
}

extern "C" void kernel_launch(void* const* d_in, const int* in_sizes, int n_in,
                              void* d_out, int out_size, void* d_ws, size_t ws_size,
                              hipStream_t stream) {
    const float* x = (const float*)d_in[0];
    char* ws = (char*)d_ws;
    // ws layout: xn16 (4 MB) | nn_idx (128 KB) | partial (512 B)
    _Float16* xn16   = (_Float16*)ws;
    int*      nn_idx = (int*)(ws + (size_t)C_N * A_N * F_N * sizeof(_Float16));
    float*    partial = (float*)(ws + (size_t)C_N * A_N * F_N * sizeof(_Float16)
                                    + (size_t)C_N * A_N * sizeof(int));
    float* out = (float*)d_out;

    k_normalize<<<(A_N * C_N) / 8, 256, 0, stream>>>(x, xn16);
    k_argmax<<<(C_N * (A_N / 16)) / 4, 128, 0, stream>>>(xn16, nn_idx);
    k_loss<<<(A_N * C_N) / 256, 256, 0, stream>>>(x, nn_idx, partial);
    k_final<<<1, 128, 0, stream>>>(partial, out);
}
